// SetConvLayer_56642028699928
// MI455X (gfx1250) — compile-verified, matrix operand order checked by reading, not run on hardware
//
#include <hip/hip_runtime.h>

// ---------------- problem constants ----------------
constexpr int BB    = 8;     // batches
constexpr int NN    = 8192;  // points
constexpr int CC    = 64;    // feature channels
constexpr int SS    = 2048;  // npoint = N * 0.25
constexpr int KSAMP = 32;    // nsample
constexpr float RR2 = 0.25f; // radius^2

constexpr int CIN = 67;      // 3 + 64 input channels to MLP
constexpr int KP0 = 96;      // CIN padded to multiple of 32
constexpr int H1  = 64;      // layer0/1 out channels
constexpr int H2  = 128;     // layer2 out channels

constexpr int P0  = 104;     // LDS pitch (halves) for XT  (96 + 8 pad, keeps 16B align)
constexpr int P1  = 72;      // LDS pitch (halves) for act (64 + 8 pad)

typedef __attribute__((ext_vector_type(16))) _Float16 v16h;
typedef __attribute__((ext_vector_type(8)))  _Float16 v8h;
typedef __attribute__((ext_vector_type(8)))  float    v8f;

// Assemble a 16-half WMMA fragment: per-lane halves {kb..kb+7, kb+16..kb+23}
// exactly as the CDNA5 16-bit A/B operand layout prescribes.
__device__ __forceinline__ v16h ld_frag(const _Float16* p) {
  v8h lo = *(const v8h*)(p);
  v8h hi = *(const v8h*)(p + 16);
  v16h r;
#pragma unroll
  for (int i = 0; i < 8; ++i) { r[i] = lo[i]; r[i + 8] = hi[i]; }
  return r;
}

// ---------------- kernel 0: convert weights f32 -> padded f16 ----------------
__global__ void cvt_weights(const float* __restrict__ w0,
                            const float* __restrict__ w1,
                            const float* __restrict__ w2,
                            _Float16* __restrict__ w0h,
                            _Float16* __restrict__ w1h,
                            _Float16* __restrict__ w2h) {
  int i = blockIdx.x * blockDim.x + threadIdx.x;
  if (i < H1 * KP0) {
    int r = i / KP0, c = i - r * KP0;
    w0h[i] = (_Float16)(c < CIN ? w0[r * CIN + c] : 0.f);
  } else if (i < H1 * KP0 + H1 * H1) {
    int j = i - H1 * KP0;
    w1h[j] = (_Float16)w1[j];
  } else if (i < H1 * KP0 + H1 * H1 + H2 * H1) {
    int j = i - H1 * KP0 - H1 * H1;
    w2h[j] = (_Float16)w2[j];
  }
}

// ---------------- kernel 1: furthest point sampling ----------------
// One workgroup per batch; each thread caches 8 points in registers.
__global__ void __launch_bounds__(1024) fps_kernel(const float* __restrict__ pos,
                                                   float* __restrict__ nxw,
                                                   float* __restrict__ out) {
  const int b = blockIdx.x;
  const int t = threadIdx.x;
  const int lane = t & 31, wid = t >> 5;

  float px[8], py[8], pz[8], dd[8];
#pragma unroll
  for (int i = 0; i < 8; ++i) {
    const float* p = pos + (size_t)(b * NN + t * 8 + i) * 3;
    px[i] = p[0]; py[i] = p[1]; pz[i] = p[2];
    dd[i] = 1e10f;
  }

  __shared__ float redv[32];
  __shared__ int   redi[32];
  __shared__ float cur[3];

  if (t == 0) {
    const float* p = pos + (size_t)(b * NN) * 3;
    cur[0] = p[0]; cur[1] = p[1]; cur[2] = p[2];
    size_t o = (size_t)(b * SS) * 3;
    out[o] = nxw[o] = p[0]; out[o+1] = nxw[o+1] = p[1]; out[o+2] = nxw[o+2] = p[2];
  }
  __syncthreads();

  for (int step = 1; step < SS; ++step) {
    const float cx = cur[0], cy = cur[1], cz = cur[2];
    float bv = -1.f; int bi = 0;
#pragma unroll
    for (int i = 0; i < 8; ++i) {
      float dx = px[i] - cx, dy = py[i] - cy, dz = pz[i] - cz;
      float d = dx * dx + dy * dy + dz * dz;
      dd[i] = fminf(dd[i], d);
      if (dd[i] > bv) { bv = dd[i]; bi = t * 8 + i; }   // ties -> smallest index
    }
#pragma unroll
    for (int off = 16; off >= 1; off >>= 1) {
      float ov = __shfl_down(bv, off);
      int   oi = __shfl_down(bi, off);
      if (ov > bv || (ov == bv && oi < bi)) { bv = ov; bi = oi; }
    }
    if (lane == 0) { redv[wid] = bv; redi[wid] = bi; }
    __syncthreads();
    if (wid == 0) {
      bv = redv[lane]; bi = redi[lane];
#pragma unroll
      for (int off = 16; off >= 1; off >>= 1) {
        float ov = __shfl_down(bv, off);
        int   oi = __shfl_down(bi, off);
        if (ov > bv || (ov == bv && oi < bi)) { bv = ov; bi = oi; }
      }
      if (lane == 0) {
        const float* p = pos + (size_t)(b * NN + bi) * 3;
        float x = p[0], y = p[1], z = p[2];
        cur[0] = x; cur[1] = y; cur[2] = z;
        size_t o = (size_t)(b * SS + step) * 3;
        out[o] = nxw[o] = x; out[o+1] = nxw[o+1] = y; out[o+2] = nxw[o+2] = z;
      }
    }
    __syncthreads();
  }
}

// ---------------- kernel 2: ball query ----------------
// One wave per center; ordered scan reproduces top_k-of-(-indices) semantics.
__global__ void __launch_bounds__(256) ballquery_kernel(const float* __restrict__ pos,
                                                        const float* __restrict__ nxw,
                                                        int* __restrict__ outidx) {
  const int lane = threadIdx.x & 31;
  const int wid  = threadIdx.x >> 5;
  const int g    = blockIdx.x * 8 + wid;      // 0..B*S-1
  const int b    = g >> 11;                   // S == 2048
  __shared__ int buf[8][KSAMP];

  const float cx = nxw[(size_t)g * 3 + 0];
  const float cy = nxw[(size_t)g * 3 + 1];
  const float cz = nxw[(size_t)g * 3 + 2];

  int cnt = 0;
  for (int n0 = 0; n0 < NN && cnt < KSAMP; n0 += 32) {
    int n = n0 + lane;
    const float* p = pos + (size_t)(b * NN + n) * 3;
    float dx = p[0] - cx, dy = p[1] - cy, dz = p[2] - cz;
    bool hit = (dx * dx + dy * dy + dz * dz) < RR2;
    unsigned m = (unsigned)__ballot(hit);
    if (hit) {
      int slot = cnt + __popc(m & ((1u << lane) - 1u));
      if (slot < KSAMP) buf[wid][slot] = n;
    }
    cnt += __popc(m);
  }
  int v;
  if (cnt == 0) v = 0;
  else {
    int c = cnt < KSAMP ? cnt : KSAMP;
    int first = buf[wid][0];
    v = (lane < c) ? buf[wid][lane] : first;
  }
  outidx[(size_t)g * KSAMP + lane] = v;
}

// ---------------- kernel 3: fused gather + 3-layer MLP (WMMA) + max-pool ----
// One workgroup (4 waves) per 2 centers = 64 GEMM columns.
__global__ void __launch_bounds__(128) setconv_mlp(
    const float* __restrict__ pos, const float* __restrict__ feat,
    const int* __restrict__ ballidx, const float* __restrict__ nxw,
    const _Float16* __restrict__ w0h, const _Float16* __restrict__ w1h,
    const _Float16* __restrict__ w2h,
    const float* __restrict__ b0, const float* __restrict__ b1,
    const float* __restrict__ b2, float* __restrict__ outf) {
  __shared__ __align__(16) _Float16 XT [64 * P0];   // input cols, K-major
  __shared__ __align__(16) _Float16 A1T[64 * P1];   // layer0 act, K-major
  __shared__ __align__(16) _Float16 A2T[64 * P1];   // layer1 act, K-major
  __shared__ float biasl[256];
  __shared__ float pmax[2][2][H2];
  __shared__ int   idxl[64];
  __shared__ float cxyz[2][3];

  const int tid = threadIdx.x;
  const int cg  = blockIdx.x * 2;     // first global center
  const int b   = cg / SS;
  const int s0  = cg % SS;

  if (tid < 64) idxl[tid] = ballidx[(size_t)(b * SS + s0 + (tid >> 5)) * KSAMP + (tid & 31)];
  if (tid < 6)  cxyz[tid / 3][tid % 3] = nxw[(size_t)(b * SS + s0 + tid / 3) * 3 + tid % 3];
  for (int i = tid; i < 256; i += 128)
    biasl[i] = (i < 64) ? b0[i] : (i < 128) ? b1[i - 64] : b2[i - 128];
  __syncthreads();

  // gather + f32->f16 into XT[col][ch], zero padded to KP0
  for (int i = tid; i < 64 * KP0; i += 128) {
    int col = i / KP0, ch = i - col * KP0;
    int n = idxl[col];
    float v = 0.f;
    if (ch < 3)        v = pos[(size_t)(b * NN + n) * 3 + ch] - cxyz[col >> 5][ch];
    else if (ch < CIN) v = feat[(size_t)b * CC * NN + (size_t)(ch - 3) * NN + n];
    XT[col * P0 + ch] = (_Float16)v;
  }
  __syncthreads();

  const int w  = tid >> 5;    // wave = column tile (16 cols)
  const int L  = tid & 31;
  const int lr = L & 15;      // row/col within tile
  const int lh = L >> 4;      // lane half selects K-offset / M-offset

  // ---- layer 0: (64 x 96) x (96 x 64) ----
  v8f zero = {};
  v8f acc0[4];
#pragma unroll
  for (int rt = 0; rt < 4; ++rt) acc0[rt] = zero;
#pragma unroll
  for (int ks = 0; ks < 3; ++ks) {
    v16h Bf = ld_frag(&XT[(w * 16 + lr) * P0 + ks * 32 + lh * 8]);
#pragma unroll
    for (int rt = 0; rt < 4; ++rt) {
      v16h Af = ld_frag(w0h + (rt * 16 + lr) * KP0 + ks * 32 + lh * 8);
      acc0[rt] = __builtin_amdgcn_wmma_f32_16x16x32_f16(false, Af, false, Bf,
                                                        (short)0, acc0[rt], false, false);
    }
  }
  // packed epilogue: 8 rows per lane are contiguous in K-major layout -> one b128 store
#pragma unroll
  for (int rt = 0; rt < 4; ++rt) {
    v8h hv;
#pragma unroll
    for (int j = 0; j < 8; ++j) {
      int m = rt * 16 + lh * 8 + j;
      float v = acc0[rt][j] + biasl[m];
      hv[j] = (_Float16)(v > 0.f ? v : 0.f);
    }
    *(v8h*)&A1T[(w * 16 + lr) * P1 + rt * 16 + lh * 8] = hv;
  }
  __syncthreads();

  // ---- layer 1: (64 x 64) x (64 x 64) ----
  v8f acc1[4];
#pragma unroll
  for (int rt = 0; rt < 4; ++rt) acc1[rt] = zero;
#pragma unroll
  for (int ks = 0; ks < 2; ++ks) {
    v16h Bf = ld_frag(&A1T[(w * 16 + lr) * P1 + ks * 32 + lh * 8]);
#pragma unroll
    for (int rt = 0; rt < 4; ++rt) {
      v16h Af = ld_frag(w1h + (rt * 16 + lr) * H1 + ks * 32 + lh * 8);
      acc1[rt] = __builtin_amdgcn_wmma_f32_16x16x32_f16(false, Af, false, Bf,
                                                        (short)0, acc1[rt], false, false);
    }
  }
#pragma unroll
  for (int rt = 0; rt < 4; ++rt) {
    v8h hv;
#pragma unroll
    for (int j = 0; j < 8; ++j) {
      int m = rt * 16 + lh * 8 + j;
      float v = acc1[rt][j] + biasl[64 + m];
      hv[j] = (_Float16)(v > 0.f ? v : 0.f);
    }
    *(v8h*)&A2T[(w * 16 + lr) * P1 + rt * 16 + lh * 8] = hv;
  }
  __syncthreads();

  // ---- layer 2: (128 x 64) x (64 x 64), fused ReLU + max over 16 cols ----
  v8f acc2[8];
#pragma unroll
  for (int rt = 0; rt < 8; ++rt) acc2[rt] = zero;
#pragma unroll
  for (int ks = 0; ks < 2; ++ks) {
    v16h Bf = ld_frag(&A2T[(w * 16 + lr) * P1 + ks * 32 + lh * 8]);
#pragma unroll
    for (int rt = 0; rt < 8; ++rt) {
      v16h Af = ld_frag(w2h + (rt * 16 + lr) * H1 + ks * 32 + lh * 8);
      acc2[rt] = __builtin_amdgcn_wmma_f32_16x16x32_f16(false, Af, false, Bf,
                                                        (short)0, acc2[rt], false, false);
    }
  }
#pragma unroll
  for (int rt = 0; rt < 8; ++rt)
#pragma unroll
    for (int j = 0; j < 8; ++j) {
      int m = rt * 16 + lh * 8 + j;
      float v = acc2[rt][j] + biasl[128 + m];
      v = v > 0.f ? v : 0.f;
#pragma unroll
      for (int off = 1; off < 16; off <<= 1) v = fmaxf(v, __shfl_xor(v, off));
      if (lr == 0) pmax[w >> 1][w & 1][m] = v;  // center (w/2), half (w&1)
    }
  __syncthreads();

  // combine the two 16-col halves of each center; write (B,128,S)
  for (int i = tid; i < 2 * H2; i += 128) {
    int ci = i >> 7, m = i & 127;
    float v = fmaxf(pmax[ci][0][m], pmax[ci][1][m]);
    outf[((size_t)b * H2 + m) * SS + (s0 + ci)] = v;
  }
}

// ---------------- launcher ----------------
extern "C" void kernel_launch(void* const* d_in, const int* in_sizes, int n_in,
                              void* d_out, int out_size, void* d_ws, size_t ws_size,
                              hipStream_t stream) {
  const float* pos  = (const float*)d_in[0];
  const float* feat = (const float*)d_in[1];
  const float* w0   = (const float*)d_in[2];
  const float* b0   = (const float*)d_in[3];
  const float* w1   = (const float*)d_in[4];
  const float* b1   = (const float*)d_in[5];
  const float* w2   = (const float*)d_in[6];
  const float* b2   = (const float*)d_in[7];
  float* out = (float*)d_out;

  char* ws = (char*)d_ws;
  _Float16* w0h   = (_Float16*)(ws);                       // 64*96*2  = 12288
  _Float16* w1h   = (_Float16*)(ws + 12288);               // 64*64*2  =  8192
  _Float16* w2h   = (_Float16*)(ws + 20480);               // 128*64*2 = 16384
  float*    nxw   = (float*)(ws + 36864);                  // B*S*3*4  = 196608
  int*      bidx  = (int*)(ws + 36864 + 196608);           // B*S*32*4 = 2097152

  cvt_weights<<<72, 256, 0, stream>>>(w0, w1, w2, w0h, w1h, w2h);
  fps_kernel<<<BB, 1024, 0, stream>>>(pos, nxw, out);
  ballquery_kernel<<<(BB * SS) / 8, 256, 0, stream>>>(pos, nxw, bidx);
  setconv_mlp<<<(BB * SS) / 2, 128, 0, stream>>>(pos, feat, bidx, nxw,
                                                 w0h, w1h, w2h, b0, b1, b2,
                                                 out + (size_t)BB * SS * 3);
}
